// MHA_20985210208688
// MI455X (gfx1250) — matrix operand order repaired and checked by reading, not compiled, not run on hardware
//
#include <hip/hip_runtime.h>
#include <hip/hip_bf16.h>
#include <math.h>
#include <stdint.h>

typedef __attribute__((ext_vector_type(16))) __bf16 v16bf;
typedef __attribute__((ext_vector_type(8)))  float  v8f;
typedef __attribute__((ext_vector_type(4)))  unsigned int u32x4;
typedef __attribute__((ext_vector_type(8)))  int i32x8;
typedef __attribute__((ext_vector_type(4)))  int i32x4;

#define BB 2
#define SS 2048
#define DD 2048
#define HH 16
#define HDm 128
#define NROW (BB*SS)     /* 4096 */
#define NQKV (3*DD)      /* 6144 */
#define BK 32
#define LP 8             /* LDS row pad (shorts) = 16B = 4 DWORDs */

__device__ __forceinline__ unsigned short f2bf_raw(float f) {
  union { float f; unsigned u; } c; c.f = f;
  unsigned u = c.u;
  unsigned r = u + 0x7FFFu + ((u >> 16) & 1u);
  return (unsigned short)(r >> 16);
}

__device__ __forceinline__ v8f vzero8() {
  v8f z;
  for (int e = 0; e < 8; ++e) z[e] = 0.0f;
  return z;
}

__device__ __forceinline__ v8f wmma_bf16(v16bf a, v16bf b, v8f c) {
  return __builtin_amdgcn_wmma_f32_16x16x32_bf16(false, a, false, b, (short)0, c,
                                                 false, false);
}

union Frag { v16bf v; uint4 u[2]; unsigned short s[16]; };

// TDM: async DMA of a 2D bf16 tile (tile0 x tile1 elements, row stride
// stride0 elements) from global to LDS, with hardware LDS row padding of
// 4 DWORDs every 2^(padIv+1) DWORDs. Tracked by TENSORcnt.
__device__ __forceinline__ void tdm_load_2d_bf16(const unsigned short* gptr,
                                                 unsigned lds_byte,
                                                 unsigned tile0, unsigned tile1,
                                                 unsigned long long stride0,
                                                 unsigned padIv) {
  unsigned long long ga = (unsigned long long)(uintptr_t)gptr;
  u32x4 g0;
  g0[0] = 1u;                                            // count=1 (valid D#)
  g0[1] = lds_byte;                                      // lds_addr
  g0[2] = (unsigned)ga;                                  // global_addr[31:0]
  g0[3] = (unsigned)((ga >> 32) & 0x1FFFFFFu) | (2u << 30); // addr[56:32]|type=2
  i32x8 g1;
  // data_size=1 (2B), pad_enable, pad_interval=padIv, pad_amount=3 (4 DWORDs)
  g1[0] = (int)((1u << 16) | (1u << 20) | (padIv << 22) | (3u << 25));
  g1[1] = (int)((tile0 & 0xFFFFu) << 16);                 // tensor_dim0 lo
  g1[2] = (int)(((tile0 >> 16) & 0xFFFFu) | ((tile1 & 0xFFFFu) << 16));
  g1[3] = (int)(((tile1 >> 16) & 0xFFFFu) | ((tile0 & 0xFFFFu) << 16)); // tile_dim0
  g1[4] = (int)(tile1 & 0xFFFFu);                         // tile_dim1 (tile_dim2=0)
  g1[5] = (int)(unsigned)(stride0 & 0xFFFFFFFFu);         // dim0 stride lo
  g1[6] = (int)(unsigned)((stride0 >> 32) & 0xFFFFu);     // dim0 stride hi
  g1[7] = 0;
  i32x4 z4; z4[0] = 0; z4[1] = 0; z4[2] = 0; z4[3] = 0;
  i32x8 z8;
  for (int i = 0; i < 8; ++i) z8[i] = 0;
  __builtin_amdgcn_tensor_load_to_lds(g0, g1, z4, z4, z8, 0);
}

// ---------------------------------------------------------------- converts
__global__ void cvt_f32_bf16(const float* __restrict__ in,
                             unsigned short* __restrict__ out, int n) {
  int i = blockIdx.x * blockDim.x + threadIdx.x;
  if (i < n) out[i] = f2bf_raw(in[i]);
}

// in: f32 [K][N] row-major  ->  out: bf16 [N][K]
__global__ void cvt_transpose(const float* __restrict__ in,
                              unsigned short* __restrict__ out, int K, int N) {
  int i = blockIdx.x * blockDim.x + threadIdx.x;
  if (i < K * N) {
    int k = i / N, n = i - k * N;
    out[(size_t)n * K + k] = f2bf_raw(in[i]);
  }
}

// ------------------------------------------------------- QKV GEMM + RoPE
__global__ __launch_bounds__(256)
void qkv_gemm(const unsigned short* __restrict__ xb,
              const unsigned short* __restrict__ wT,
              const float* __restrict__ bias,
              unsigned short* __restrict__ Qb,
              unsigned short* __restrict__ Kb,
              unsigned short* __restrict__ Vb) {
  __shared__ unsigned short Ash[2][128][BK + LP];   // [m][k], double-buffered
  __shared__ unsigned short Bsh[2][128][BK + LP];   // [n][k]
  const int tid  = threadIdx.x;
  const int lane = tid & 31;
  const int wave = tid >> 5;
  const int rowBase = blockIdx.y * 128;
  const int colBase = blockIdx.x * 128;

  v8f acc[8];
  for (int i = 0; i < 8; ++i) acc[i] = vzero8();

  const int nk = DD / BK;   // 64 k-tiles
  if (wave == 0) {          // prologue: stream tile 0 via TDM
    tdm_load_2d_bf16(&xb[(size_t)rowBase * DD], (unsigned)(uintptr_t)&Ash[0][0][0],
                     BK, 128, DD, 3);
    tdm_load_2d_bf16(&wT[(size_t)colBase * DD], (unsigned)(uintptr_t)&Bsh[0][0][0],
                     BK, 128, DD, 3);
  }

  for (int it = 0; it < nk; ++it) {
    const int buf = it & 1;
    __syncthreads();                          // buffer (it+1)&1 fully consumed
    if (wave == 0) {
      if (it + 1 < nk) {
        int kt = (it + 1) * BK;
        int nb = (it + 1) & 1;
        tdm_load_2d_bf16(&xb[(size_t)rowBase * DD + kt],
                         (unsigned)(uintptr_t)&Ash[nb][0][0], BK, 128, DD, 3);
        tdm_load_2d_bf16(&wT[(size_t)colBase * DD + kt],
                         (unsigned)(uintptr_t)&Bsh[nb][0][0], BK, 128, DD, 3);
        __builtin_amdgcn_s_wait_tensorcnt(2); // tile it complete, it+1 in flight
      } else {
        __builtin_amdgcn_s_wait_tensorcnt(0);
      }
    }
    __syncthreads();                          // tile it visible to all waves

    Frag afr;                                 // A: 16x32, M = lane%16
    int arow = wave * 16 + (lane & 15);
    int ak   = (lane >> 4) * 8;
    afr.u[0] = *(const uint4*)&Ash[buf][arow][ak];
    afr.u[1] = *(const uint4*)&Ash[buf][arow][16 + ak];
    for (int nt = 0; nt < 8; ++nt) {          // B: 32x16, N = lane%16
      Frag bfr;
      int bn = nt * 16 + (lane & 15);
      int bk = (lane >> 4) * 8;
      bfr.u[0] = *(const uint4*)&Bsh[buf][bn][bk];
      bfr.u[1] = *(const uint4*)&Bsh[buf][bn][bk + 16];
      acc[nt] = wmma_bf16(afr.v, bfr.v, acc[nt]);
    }
  }

  // epilogue: bias (+RoPE for q,k), scatter to [B][H][S][HD] bf16
  const int ct    = blockIdx.x;       // 0..47: q heads 0-15, k 16-31, v 32-47
  const int which = ct >> 4;
  const int h     = ct & 15;
  const int col0  = lane & 15;
  const int rowLocal = wave * 16 + (lane >> 4) * 8;

  float bcol[8];
  for (int n = 0; n < 8; ++n) bcol[n] = bias[colBase + n * 16 + col0];
  unsigned short* dst = (which == 0) ? Qb : (which == 1) ? Kb : Vb;

  for (int r = 0; r < 8; ++r) {
    int grow = rowBase + rowLocal + r;
    int b = grow / SS, s = grow - b * SS;
    float vals[8];
    for (int n = 0; n < 8; ++n) vals[n] = acc[n][r] + bcol[n];
    if (which < 2) {                          // RoPE over full head (ROT=128)
      for (int n = 0; n < 4; ++n) {
        int j = n * 16 + col0;                // 0..63
        float invf = expf((float)j * -0.14391156831333678f); // -ln(1e4)/64
        float th = (float)s * invf;
        float cth = cosf(th), sth = sinf(th);
        float x1 = vals[n], x2 = vals[n + 4];
        vals[n]     = x1 * cth - x2 * sth;
        vals[n + 4] = x2 * cth + x1 * sth;
      }
    }
    size_t base = ((size_t)(b * HH + h) * SS + s) * HDm;
    for (int n = 0; n < 8; ++n)
      dst[base + n * 16 + col0] = f2bf_raw(vals[n]);
  }
}

// ------------------------------------------------------- flash attention
__global__ __launch_bounds__(128)
void attn(const unsigned short* __restrict__ Qb,
          const unsigned short* __restrict__ Kb,
          const unsigned short* __restrict__ Vb,
          unsigned short* __restrict__ Cb) {
  __shared__ unsigned short Ksh[32][HDm + LP];    // [key][hd] (TDM-padded rows)
  __shared__ unsigned short VshT[HDm][32 + LP];   // [hd][key] manual transpose
  __shared__ unsigned short Psh[4][16][32 + LP];  // per-wave P tile
  const int tid  = threadIdx.x;
  const int lane = tid & 31;
  const int wave = tid >> 5;                      // 0..3, each owns 16 q-rows
  const int qt = blockIdx.x;
  const int h  = blockIdx.y;
  const int b  = blockIdx.z;
  const int qBase = qt * 64;
  const size_t headOff = (size_t)(b * HH + h) * SS * HDm;

  Frag qf[4];                                     // Q A-frags, HD=128
  {
    int qrow = qBase + wave * 16 + (lane & 15);
    const unsigned short* qp = Qb + headOff + (size_t)qrow * HDm;
    for (int kk = 0; kk < 4; ++kk) {
      int k0 = kk * 32 + (lane >> 4) * 8;
      qf[kk].u[0] = *(const uint4*)&qp[k0];
      qf[kk].u[1] = *(const uint4*)&qp[k0 + 16];
    }
  }

  v8f o[8];
  for (int i = 0; i < 8; ++i) o[i] = vzero8();
  float m[8], lsum[8];
  for (int r = 0; r < 8; ++r) { m[r] = -3.0e38f; lsum[r] = 0.0f; }

  const float scale = 0.08838834764831845f;       // 1/sqrt(128)
  const int nTiles = qt * 2 + 2;                  // causal extent
  const int myrow0 = qBase + wave * 16 + (lane >> 4) * 8;

  for (int jt = 0; jt < nTiles; ++jt) {
    const int kB = jt * 32;
    __syncthreads();                              // prev tile fully consumed
    if (wave == 0)                                // K tile 32x128 via TDM
      tdm_load_2d_bf16(Kb + headOff + (size_t)kB * HDm,
                       (unsigned)(uintptr_t)&Ksh[0][0], HDm, 32, HDm, 5);
    {                                             // V tile transposed, manual
      uint4 vd[4];
      for (int l = 0; l < 4; ++l) {
        int c = tid + 128 * l;
        int kr = c >> 4, cc = (c & 15) * 8;
        vd[l] = *(const uint4*)&Vb[headOff + (size_t)(kB + kr) * HDm + cc];
      }
      for (int l = 0; l < 4; ++l) {
        int c = tid + 128 * l;
        int kr = c >> 4, cc = (c & 15) * 8;
        const unsigned short* ds = (const unsigned short*)&vd[l];
        for (int e = 0; e < 8; ++e) VshT[cc + e][kr] = ds[e];
      }
    }
    if (wave == 0) __builtin_amdgcn_s_wait_tensorcnt(0);
    __syncthreads();

    v8f sf[2];                                    // S = Q K^T (two 16-key halves)
    for (int nh = 0; nh < 2; ++nh) {
      v8f accs = vzero8();
      int key = nh * 16 + (lane & 15);
      for (int kk = 0; kk < 4; ++kk) {
        Frag bf;
        int k0 = kk * 32 + (lane >> 4) * 8;
        bf.u[0] = *(const uint4*)&Ksh[key][k0];
        bf.u[1] = *(const uint4*)&Ksh[key][k0 + 16];
        accs = wmma_bf16(qf[kk].v, bf.v, accs);
      }
      sf[nh] = accs;
    }
    for (int nh = 0; nh < 2; ++nh) {              // scale + causal mask
      int key = kB + nh * 16 + (lane & 15);
      for (int r = 0; r < 8; ++r) {
        float sv = sf[nh][r] * scale;
        sf[nh][r] = (key <= myrow0 + r) ? sv : -3.0e38f;
      }
    }
    for (int r = 0; r < 8; ++r) {                 // online softmax per row
      float t = fmaxf(sf[0][r], sf[1][r]);
      for (int d = 1; d < 16; d <<= 1) t = fmaxf(t, __shfl_xor(t, d, 32));
      float mn = fmaxf(m[r], t);
      float corr = __expf(m[r] - mn);
      float p0 = __expf(sf[0][r] - mn);
      float p1 = __expf(sf[1][r] - mn);
      sf[0][r] = p0; sf[1][r] = p1;
      float ps = p0 + p1;
      for (int d = 1; d < 16; d <<= 1) ps += __shfl_xor(ps, d, 32);
      lsum[r] = lsum[r] * corr + ps;
      m[r] = mn;
      for (int nf = 0; nf < 8; ++nf) o[nf][r] *= corr;
    }
    for (int r = 0; r < 8; ++r) {                 // P (C-layout) -> LDS bf16
      int prow = (lane >> 4) * 8 + r;
      Psh[wave][prow][lane & 15]        = f2bf_raw(sf[0][r]);
      Psh[wave][prow][16 + (lane & 15)] = f2bf_raw(sf[1][r]);
    }
    __syncthreads();
    Frag pf;                                      // reload P as A-frag (16x32)
    {
      int pr = lane & 15;
      int k0 = (lane >> 4) * 8;
      pf.u[0] = *(const uint4*)&Psh[wave][pr][k0];
      pf.u[1] = *(const uint4*)&Psh[wave][pr][16 + k0];
    }
    for (int nf = 0; nf < 8; ++nf) {              // O += P * V
      Frag bf;
      int hd = nf * 16 + (lane & 15);
      int k0 = (lane >> 4) * 8;
      bf.u[0] = *(const uint4*)&VshT[hd][k0];
      bf.u[1] = *(const uint4*)&VshT[hd][k0 + 16];
      o[nf] = wmma_bf16(pf.v, bf.v, o[nf]);
    }
  }

  for (int r = 0; r < 8; ++r) {                   // normalize, store ctx bf16
    float inv = 1.0f / lsum[r];
    int s = qBase + wave * 16 + (lane >> 4) * 8 + r;
    size_t rowp = ((size_t)b * SS + s) * DD + h * HDm;
    for (int nf = 0; nf < 8; ++nf)
      Cb[rowp + nf * 16 + (lane & 15)] = f2bf_raw(o[nf][r] * inv);
  }
}

// ------------------------------------------------------- output projection
__global__ __launch_bounds__(256)
void out_gemm(const unsigned short* __restrict__ Cb,
              const unsigned short* __restrict__ wT,
              const float* __restrict__ bias,
              float* __restrict__ out) {
  __shared__ unsigned short Ash[2][128][BK + LP];
  __shared__ unsigned short Bsh[2][128][BK + LP];
  const int tid  = threadIdx.x;
  const int lane = tid & 31;
  const int wave = tid >> 5;
  const int rowBase = blockIdx.y * 128;
  const int colBase = blockIdx.x * 128;

  v8f acc[8];
  for (int i = 0; i < 8; ++i) acc[i] = vzero8();

  const int nk = DD / BK;
  if (wave == 0) {
    tdm_load_2d_bf16(&Cb[(size_t)rowBase * DD], (unsigned)(uintptr_t)&Ash[0][0][0],
                     BK, 128, DD, 3);
    tdm_load_2d_bf16(&wT[(size_t)colBase * DD], (unsigned)(uintptr_t)&Bsh[0][0][0],
                     BK, 128, DD, 3);
  }

  for (int it = 0; it < nk; ++it) {
    const int buf = it & 1;
    __syncthreads();
    if (wave == 0) {
      if (it + 1 < nk) {
        int kt = (it + 1) * BK;
        int nb = (it + 1) & 1;
        tdm_load_2d_bf16(&Cb[(size_t)rowBase * DD + kt],
                         (unsigned)(uintptr_t)&Ash[nb][0][0], BK, 128, DD, 3);
        tdm_load_2d_bf16(&wT[(size_t)colBase * DD + kt],
                         (unsigned)(uintptr_t)&Bsh[nb][0][0], BK, 128, DD, 3);
        __builtin_amdgcn_s_wait_tensorcnt(2);
      } else {
        __builtin_amdgcn_s_wait_tensorcnt(0);
      }
    }
    __syncthreads();

    Frag afr;
    int arow = wave * 16 + (lane & 15);
    int ak   = (lane >> 4) * 8;
    afr.u[0] = *(const uint4*)&Ash[buf][arow][ak];
    afr.u[1] = *(const uint4*)&Ash[buf][arow][16 + ak];
    for (int nt = 0; nt < 8; ++nt) {
      Frag bfr;
      int bn = nt * 16 + (lane & 15);
      int bk = (lane >> 4) * 8;
      bfr.u[0] = *(const uint4*)&Bsh[buf][bn][bk];
      bfr.u[1] = *(const uint4*)&Bsh[buf][bn][bk + 16];
      acc[nt] = wmma_bf16(afr.v, bfr.v, acc[nt]);
    }
  }

  const int col0 = lane & 15;
  const int rowLocal = wave * 16 + (lane >> 4) * 8;
  for (int r = 0; r < 8; ++r) {
    int grow = rowBase + rowLocal + r;
    for (int n = 0; n < 8; ++n) {
      int gcol = colBase + n * 16 + col0;
      out[(size_t)grow * DD + gcol] = acc[n][r] + bias[gcol];
    }
  }
}

// ---------------------------------------------------------------- launch
extern "C" void kernel_launch(void* const* d_in, const int* in_sizes, int n_in,
                              void* d_out, int out_size, void* d_ws, size_t ws_size,
                              hipStream_t stream) {
  const float* x    = (const float*)d_in[0];
  const float* wqkv = (const float*)d_in[1];
  const float* bqkv = (const float*)d_in[2];
  const float* wout = (const float*)d_in[3];
  const float* bout = (const float*)d_in[4];
  float* out = (float*)d_out;

  unsigned short* p = (unsigned short*)d_ws;
  unsigned short* xb  = p; p += (size_t)NROW * DD;
  unsigned short* wqT = p; p += (size_t)NQKV * DD;
  unsigned short* woT = p; p += (size_t)DD * DD;
  unsigned short* Qb  = p; p += (size_t)BB * HH * SS * HDm;
  unsigned short* Kb  = p; p += (size_t)BB * HH * SS * HDm;
  unsigned short* Vb  = p; p += (size_t)BB * HH * SS * HDm;
  unsigned short* Cb  = p;

  int n1 = NROW * DD;
  cvt_f32_bf16<<<(n1 + 255) / 256, 256, 0, stream>>>(x, xb, n1);
  int n2 = DD * NQKV;
  cvt_transpose<<<(n2 + 255) / 256, 256, 0, stream>>>(wqkv, wqT, DD, NQKV);
  int n3 = DD * DD;
  cvt_transpose<<<(n3 + 255) / 256, 256, 0, stream>>>(wout, woT, DD, DD);

  qkv_gemm<<<dim3(48, 32), 256, 0, stream>>>(xb, wqT, bqkv, Qb, Kb, Vb);
  attn<<<dim3(SS / 64, HH, BB), 128, 0, stream>>>(Qb, Kb, Vb, Cb);
  out_gemm<<<dim3(16, 32), 256, 0, stream>>>(Cb, woT, bout, out);
}
